// SpectralVideoHead_51410758533293
// MI455X (gfx1250) — matrix operand order repaired and checked by reading, not compiled, not run on hardware
//
#include <hip/hip_runtime.h>
#include <hip/hip_bf16.h>

// ---------------- problem dims ----------------
#define BV   8
#define TT   16
#define CC   3
#define HH   224
#define WW   224
#define NPAT 196          // (224/16)^2
#define DD   512
#define DCC  8
#define DFF  16
#define KK   64
#define PDIM 768          // C*P*P
#define MROWS (BV*TT*NPAT)   // 25088
#define MBNP  (BV*NPAT)      // 1568

// ALPHA = 0.7 - (540-10)/(2500-10)*0.4
__device__ __constant__ float kALPHA = 0.61485943775100401f;

// scan coefficients: g = exp(-1/0.5)*cos(2*pi/2) = -e^-2 ; w_k = exp(-k/5)
#define GDEC  (-0.13533528323661270f)
#define FW1   (0.81873075307798182f)
#define FW2   (0.67032004603563930f)
#define FW3   (0.54881163609402643f)
#define FW4   (0.44932896411722156f)

typedef __attribute__((ext_vector_type(16))) __bf16 v16bf;
typedef __attribute__((ext_vector_type(8)))  float  v8f;

union BFrag { v16bf v; __bf16 e[16]; uint4 q[2]; };
union CFrag { v8f  v; float  e[8]; };

// =====================================================================
// K0: weight convert+transpose  Wt[n][k] = bf16(W[k][n])
// one block per output column n
// =====================================================================
__global__ __launch_bounds__(256) void k_wcvt(
    const float* __restrict__ W, __bf16* __restrict__ Wt, int Kd)
{
    const int n = blockIdx.x;                   // 0..DD-1
    for (int k = threadIdx.x; k < Kd; k += 256)
        Wt[(long)n * Kd + k] = (__bf16)W[(long)k * DD + n];
}

// =====================================================================
// K1: patchify + X = pat @ W_patch + b_patch + pos_embed   (WMMA bf16)
// grid (1568, 16), block 32: one wave -> 16x32 output tile (2 acc, A reuse)
// A fed by float4 vector loads (contiguous pixel runs), B by b128 bf16 loads
// =====================================================================
__global__ __launch_bounds__(32) void k_patch_embed(
    const float* __restrict__ video, const __bf16* __restrict__ Wt,
    const float* __restrict__ b_patch, const float* __restrict__ pos_embed,
    float* __restrict__ xout)
{
    const int mtile = blockIdx.x;
    const int npair = blockIdx.y;
    const int lane  = threadIdx.x;
    const int rowin = lane & 15;
    const int akb   = (lane >> 4) * 8;    // A-frag K base per half-wave
    const int bkb   = (lane >> 4) * 8;    // B-frag K base per half-wave
    const int ncol0 = npair * 32 + (lane & 15);
    const int ncol1 = ncol0 + 16;

    // row -> video gather base (patchify on the fly)
    const int  mrow = mtile * 16 + rowin;
    const int  bt   = mrow / NPAT;
    const int  np   = mrow - bt * NPAT;
    const int  ph   = np / 14;
    const int  pw   = np - ph * 14;
    const long rowbase = (long)bt * (CC * HH * WW) + (long)(ph * 16) * WW + pw * 16;

    const __bf16* w0 = Wt + (long)ncol0 * PDIM + bkb;
    const __bf16* w1 = Wt + (long)ncol1 * PDIM + bkb;

    CFrag c0, c1;
#pragma unroll
    for (int r = 0; r < 8; ++r) { c0.e[r] = 0.0f; c1.e[r] = 0.0f; }

    for (int kk = 0; kk < PDIM / 32; ++kk) {
        BFrag a, b0, b1;
        // A: two groups of 8 K-contiguous elements -> 2 x float4 each
#pragma unroll
        for (int g = 0; g < 2; ++g) {
            const int k0 = kk * 32 + g * 16 + akb;     // (k0 & 15) == akb
            const long off = (long)(k0 >> 8) * (HH * WW) + ((k0 >> 4) & 15) * WW + akb;
            const float4* p = reinterpret_cast<const float4*>(video + rowbase + off);
            const float4 f0 = p[0], f1 = p[1];
            a.e[8*g+0] = (__bf16)f0.x; a.e[8*g+1] = (__bf16)f0.y;
            a.e[8*g+2] = (__bf16)f0.z; a.e[8*g+3] = (__bf16)f0.w;
            a.e[8*g+4] = (__bf16)f1.x; a.e[8*g+5] = (__bf16)f1.y;
            a.e[8*g+6] = (__bf16)f1.z; a.e[8*g+7] = (__bf16)f1.w;
        }
        // B: two groups of 8 K-contiguous bf16 from transposed weights -> 2 x b128
        b0.q[0] = *reinterpret_cast<const uint4*>(w0 + kk * 32);
        b0.q[1] = *reinterpret_cast<const uint4*>(w0 + kk * 32 + 16);
        b1.q[0] = *reinterpret_cast<const uint4*>(w1 + kk * 32);
        b1.q[1] = *reinterpret_cast<const uint4*>(w1 + kk * 32 + 16);
        if (kk + 2 < PDIM / 32) {
            __builtin_prefetch(w0 + (kk + 2) * 32, 0, 0);   // global_prefetch_b8
            __builtin_prefetch(w1 + (kk + 2) * 32, 0, 0);
        }
        c0.v = __builtin_amdgcn_wmma_f32_16x16x32_bf16(
                   false, a.v, false, b0.v, (short)0, c0.v, false, false);
        c1.v = __builtin_amdgcn_wmma_f32_16x16x32_bf16(
                   false, a.v, false, b1.v, (short)0, c1.v, false, false);
    }

    const int np0 = (mtile * 16) % NPAT;   // cheap wrap instead of 8 magic-divs
    const float bp0 = b_patch[ncol0], bp1 = b_patch[ncol1];
#pragma unroll
    for (int r = 0; r < 8; ++r) {
        const int row = (lane < 16) ? r : (r + 8);
        const int mg  = mtile * 16 + row;
        int npg = np0 + row; if (npg >= NPAT) npg -= NPAT;
        xout[(long)mg * DD + ncol0] = c0.e[r] + bp0 + pos_embed[npg * DD + ncol0];
        xout[(long)mg * DD + ncol1] = c1.e[r] + bp1 + pos_embed[npg * DD + ncol1];
    }
}

// =====================================================================
// K2: phase = ALPHA*sum(tanh(x @ W_phase)); xf = x*cos(phase)
// one wave per row; grid 3136, block 256 (8 waves)
// =====================================================================
__global__ __launch_bounds__(256) void k_phase(
    const float* __restrict__ x, const float* __restrict__ W_phase,
    float* __restrict__ xf, float* __restrict__ phases)
{
    const int wid  = threadIdx.x >> 5;
    const int lane = threadIdx.x & 31;
    const int m    = blockIdx.x * 8 + wid;
    const float* xr = x + (long)m * DD;

    float acc[DCC];
#pragma unroll
    for (int j = 0; j < DCC; ++j) acc[j] = 0.0f;

    for (int d = lane; d < DD; d += 32) {
        const float xv = xr[d];
#pragma unroll
        for (int j = 0; j < DCC; ++j) acc[j] += xv * W_phase[d * DCC + j];
    }
#pragma unroll
    for (int s = 16; s > 0; s >>= 1) {
#pragma unroll
        for (int j = 0; j < DCC; ++j) acc[j] += __shfl_xor(acc[j], s, 32);
    }
    float ph = 0.0f;
#pragma unroll
    for (int j = 0; j < DCC; ++j) ph += tanhf(acc[j]);
    ph *= kALPHA;
    const float cp = cosf(ph);

    for (int d = lane; d < DD; d += 32) xf[(long)m * DD + d] = xr[d] * cp;
    if (lane == 0) phases[m] = ph;
}

// =====================================================================
// K3: linear temporal recurrence (the lax.scan, parallel over (b,np,d))
// =====================================================================
__global__ __launch_bounds__(256) void k_scan(
    const float* __restrict__ xf, float* __restrict__ xmid)
{
    const int idx  = blockIdx.x * 256 + threadIdx.x;   // over B*NP*D
    const int d    = idx & (DD - 1);
    const int rest = idx >> 9;                         // b*NP + np
    const int b    = rest / NPAT;
    const int np   = rest - b * NPAT;

    const long base   = (((long)b * TT) * NPAT + np) * DD + d;
    const long stride = (long)NPAT * DD;

    float m = 0.f, h0 = 0.f, h1 = 0.f, h2 = 0.f, h3 = 0.f;
#pragma unroll
    for (int t = 0; t < TT; ++t) {
        const float xv = xf[base + t * stride];
        m = GDEC * m + xv + FW1 * h0 + FW2 * h1 + FW3 * h2 + FW4 * h3;
        h3 = h2; h2 = h1; h1 = h0; h0 = xv;
        xmid[base + t * stride] = xv + m;
    }
}

// =====================================================================
// K4: causal field layer: f = x@W_f+b_f ; depthwise conv K=64 ; gelu ;
//     y = x + gelu(fc)@W_o + b_o + temporal*tf      (per (b,t) block)
// =====================================================================
__global__ __launch_bounds__(256) void k_field(
    const float* __restrict__ xmid, const float* __restrict__ W_f,
    const float* __restrict__ b_f, const float* __restrict__ conv_k,
    const float* __restrict__ W_o, const float* __restrict__ b_o,
    const float* __restrict__ temporal, float* __restrict__ y)
{
    __shared__ float f_lds[DFF][NPAT + 4];   // [df][np]
    __shared__ float g_lds[NPAT][DFF];       // gelu(conv) [np][df]

    const int  bt = blockIdx.x;              // b*T + t
    const int  t  = bt & (TT - 1);
    const float tf = (float)t / (float)TT;
    const long rowbase = (long)bt * NPAT * DD;

    // stage 1: f[np][df]
    for (int idx = threadIdx.x; idx < NPAT * DFF; idx += 256) {
        const int np = idx >> 4, df = idx & 15;
        const float* xr = xmid + rowbase + (long)np * DD;
        float s = b_f[df];
        for (int d = 0; d < DD; ++d) s += xr[d] * W_f[d * DFF + df];
        f_lds[df][np] = s;
    }
    __syncthreads();

    // stage 2: causal depthwise conv + exact gelu
    for (int idx = threadIdx.x; idx < NPAT * DFF; idx += 256) {
        const int np = idx >> 4, df = idx & 15;
        float s = 0.0f;
#pragma unroll 8
        for (int j = 0; j < KK; ++j) {
            const int src = np + j - (KK - 1);
            if (src >= 0) s += conv_k[df * KK + j] * f_lds[df][src];
        }
        g_lds[np][df] = 0.5f * s * (1.0f + erff(s * 0.70710678118654752f));
    }
    __syncthreads();

    // stage 3: expand through W_o + residual + temporal embed
    for (int idx = threadIdx.x; idx < NPAT * DD; idx += 256) {
        const int np = idx >> 9, d = idx & (DD - 1);
        float s = b_o[d] + temporal[d] * tf + xmid[rowbase + idx];
#pragma unroll
        for (int df = 0; df < DFF; ++df) s += g_lds[np][df] * W_o[df * DD + d];
        y[rowbase + idx] = s;
    }
}

// =====================================================================
// K5: LayerNorm (fused to bf16 LDS tile) + out = xn @ W_proj + b_proj
// grid 1568, block 128 (4 waves); A via ds b128 loads, B via bf16 Wt
// =====================================================================
__global__ __launch_bounds__(128) void k_out(
    const float* __restrict__ y, const float* __restrict__ ln_g,
    const float* __restrict__ ln_b, const __bf16* __restrict__ Wt,
    const float* __restrict__ b_proj, float* __restrict__ out)
{
    __shared__ __bf16 xn[16 * DD];           // normalized tile, 16 KB
    __shared__ float red_s[16][8], red_q[16][8];
    __shared__ float mu_s[16], rs_s[16];

    const int  mtile = blockIdx.x;
    const long base  = (long)mtile * 16 * DD;
    const int  tid   = threadIdx.x;

    { // per-row stats: 8 threads/row x 64 elems
        const int row = tid >> 3, seg = tid & 7;
        float s = 0.f, q = 0.f;
        const float4* yr = reinterpret_cast<const float4*>(y + base + (long)row * DD + seg * 64);
#pragma unroll
        for (int i = 0; i < 16; ++i) {
            const float4 v = yr[i];
            s += v.x + v.y + v.z + v.w;
            q += v.x * v.x + v.y * v.y + v.z * v.z + v.w * v.w;
        }
        red_s[row][seg] = s; red_q[row][seg] = q;
    }
    __syncthreads();
    if (tid < 16) {
        float s = 0.f, q = 0.f;
#pragma unroll
        for (int j = 0; j < 8; ++j) { s += red_s[tid][j]; q += red_q[tid][j]; }
        const float mu = s * (1.0f / DD);
        mu_s[tid] = mu;
        rs_s[tid] = rsqrtf(q * (1.0f / DD) - mu * mu + 1e-5f);
    }
    __syncthreads();
    for (int idx = tid; idx < 16 * DD; idx += 128) {
        const int row = idx >> 9, d = idx & (DD - 1);
        const float v = (y[base + idx] - mu_s[row]) * rs_s[row] * ln_g[d] + ln_b[d];
        xn[idx] = (__bf16)v;
    }
    __syncthreads();

    const int wid = tid >> 5, lane = tid & 31;
    const int rowin = lane & 15;
    const int akb = (lane >> 4) * 8;
    const int bkb = (lane >> 4) * 8;

    for (int pair = wid; pair < 16; pair += 4) {   // 2 N-tiles per iter, A reuse
        const int ncol0 = pair * 32 + (lane & 15);
        const int ncol1 = ncol0 + 16;
        const __bf16* w0 = Wt + (long)ncol0 * DD + bkb;
        const __bf16* w1 = Wt + (long)ncol1 * DD + bkb;

        CFrag c0, c1;
#pragma unroll
        for (int r = 0; r < 8; ++r) { c0.e[r] = 0.0f; c1.e[r] = 0.0f; }

        for (int kk = 0; kk < DD / 32; ++kk) {
            BFrag a, b0, b1;
            a.q[0] = *reinterpret_cast<const uint4*>(&xn[rowin * DD + kk * 32 + akb]);
            a.q[1] = *reinterpret_cast<const uint4*>(&xn[rowin * DD + kk * 32 + akb + 16]);
            b0.q[0] = *reinterpret_cast<const uint4*>(w0 + kk * 32);
            b0.q[1] = *reinterpret_cast<const uint4*>(w0 + kk * 32 + 16);
            b1.q[0] = *reinterpret_cast<const uint4*>(w1 + kk * 32);
            b1.q[1] = *reinterpret_cast<const uint4*>(w1 + kk * 32 + 16);
            c0.v = __builtin_amdgcn_wmma_f32_16x16x32_bf16(
                       false, a.v, false, b0.v, (short)0, c0.v, false, false);
            c1.v = __builtin_amdgcn_wmma_f32_16x16x32_bf16(
                       false, a.v, false, b1.v, (short)0, c1.v, false, false);
        }
        const float bp0 = b_proj[ncol0], bp1 = b_proj[ncol1];
#pragma unroll
        for (int r = 0; r < 8; ++r) {
            const int row = (lane < 16) ? r : (r + 8);
            out[base + (long)row * DD + ncol0] = c0.e[r] + bp0;
            out[base + (long)row * DD + ncol1] = c1.e[r] + bp1;
        }
    }
}

// =====================================================================
extern "C" void kernel_launch(void* const* d_in, const int* in_sizes, int n_in,
                              void* d_out, int out_size, void* d_ws, size_t ws_size,
                              hipStream_t stream) {
    (void)in_sizes; (void)n_in; (void)out_size; (void)ws_size;
    const float* video     = (const float*)d_in[0];
    const float* W_patch   = (const float*)d_in[1];
    const float* b_patch   = (const float*)d_in[2];
    const float* pos_embed = (const float*)d_in[3];
    const float* temporal  = (const float*)d_in[4];
    const float* W_phase   = (const float*)d_in[5];
    const float* W_f       = (const float*)d_in[6];
    const float* b_f       = (const float*)d_in[7];
    const float* conv_k    = (const float*)d_in[8];
    const float* W_o       = (const float*)d_in[9];
    const float* b_o       = (const float*)d_in[10];
    const float* ln_g      = (const float*)d_in[11];
    const float* ln_b      = (const float*)d_in[12];
    const float* W_proj    = (const float*)d_in[13];
    const float* b_proj    = (const float*)d_in[14];

    float* out    = (float*)d_out;
    float* phases = out + (long)MROWS * DD;          // tuple output #2
    float* bufA   = (float*)d_ws;                    // x -> x_mid
    float* bufB   = bufA + (long)MROWS * DD;         // xf -> y
    __bf16* wtp   = (__bf16*)(bufB + (long)MROWS * DD);  // W_patch^T bf16 [512][768]
    __bf16* wtq   = wtp + (long)DD * PDIM;               // W_proj^T  bf16 [512][512]

    k_wcvt<<<DD, 256, 0, stream>>>(W_patch, wtp, PDIM);
    k_wcvt<<<DD, 256, 0, stream>>>(W_proj,  wtq, DD);

    dim3 g1(MROWS / 16, DD / 32);
    k_patch_embed<<<g1, 32, 0, stream>>>(video, wtp, b_patch, pos_embed, bufA);
    k_phase<<<MROWS / 8, 256, 0, stream>>>(bufA, W_phase, bufB, phases);
    k_scan<<<(MBNP * DD) / 256, 256, 0, stream>>>(bufB, bufA);
    k_field<<<BV * TT, 256, 0, stream>>>(bufA, W_f, b_f, conv_k, W_o, b_o, temporal, bufB);
    k_out<<<MROWS / 16, 128, 0, stream>>>(bufB, ln_g, ln_b, wtq, b_proj, out);
}
